// GPSAttentionLayer_55061480735235
// MI455X (gfx1250) — matrix operand order repaired and checked by reading, not compiled, not run on hardware
//
#include <hip/hip_runtime.h>

// ---------------- problem constants ----------------
static constexpr int    NN   = 20000;   // nodes
static constexpr int    FEAT = 256;     // FI == FO
static constexpr int    AE   = 16;      // attention embedding
static constexpr int    RFN  = 32;      // receptive field width
static constexpr int    TK   = 16;      // top-k stage 1
static constexpr float  ALPHA = 0.2f;
static constexpr float  EPS   = 1e-5f;

// ---------------- workspace layout (bytes) ----------------
static constexpr size_t OFF_NEWH = 0;                                  // N*256 f32
static constexpr size_t OFF_KEY  = OFF_NEWH + (size_t)NN * FEAT * 4;   // N*16 f32
static constexpr size_t OFF_QRY  = OFF_KEY  + (size_t)NN * AE * 4;     // N*16 f32
static constexpr size_t OFF_LAS  = OFF_QRY  + (size_t)NN * AE * 4;     // N f32
static constexpr size_t OFF_RAS  = OFF_LAS  + (size_t)NN * 4;          // N f32
static constexpr size_t OFF_ATT  = OFF_RAS  + (size_t)NN * 4;          // N*32 f32
static constexpr size_t OFF_XB   = OFF_ATT  + (size_t)NN * RFN * 4;    // N*256 bf16
static constexpr size_t OFF_WT   = OFF_XB   + (size_t)NN * FEAT * 2;   // 256*256 bf16 (transposed)
static constexpr size_t OFF_WKQT = OFF_WT   + (size_t)FEAT * FEAT * 2; // 32*256 bf16 (transposed)
static constexpr size_t OFF_BNS  = OFF_WKQT + (size_t)32 * FEAT * 2;   // 256 f32
static constexpr size_t OFF_BNQ  = OFF_BNS + 1024;                     // 256 f32
static constexpr size_t OFF_SC   = OFF_BNQ + 1024;                     // 256 f32
static constexpr size_t OFF_SH   = OFF_SC  + 1024;                     // 256 f32
static constexpr size_t OFF_MIN  = OFF_SH  + 1024;                     // 2 u32 (att_min, natt_min)

// ---------------- vector types ----------------
typedef __attribute__((ext_vector_type(8)))  float    v8f;
typedef __attribute__((ext_vector_type(16))) __bf16   v16bf;
typedef __attribute__((ext_vector_type(8)))  __bf16   v8bf;
typedef __attribute__((ext_vector_type(4)))  unsigned u32x4;
typedef __attribute__((ext_vector_type(8)))  unsigned u32x8;

// ---------------- small device helpers ----------------
__device__ __forceinline__ __bf16 f2bf(float f) {
    unsigned u = __float_as_uint(f);
    unsigned r = (u + 0x7FFFu + ((u >> 16) & 1u)) >> 16;   // round-nearest-even
    unsigned short s = (unsigned short)r;
    return __builtin_bit_cast(__bf16, s);
}
// order-preserving float<->uint map (for atomicMin on floats)
__device__ __forceinline__ unsigned fordU(float f) {
    unsigned u = __float_as_uint(f);
    return (u & 0x80000000u) ? ~u : (u | 0x80000000u);
}
__device__ __forceinline__ float uordF(unsigned u) {
    unsigned v = (u & 0x80000000u) ? (u & 0x7FFFFFFFu) : ~u;
    return __uint_as_float(v);
}
__device__ __forceinline__ float wredMaxF(float v) {
    #pragma unroll
    for (int m = 16; m >= 1; m >>= 1) v = fmaxf(v, __shfl_xor(v, m, 32));
    return v;
}
__device__ __forceinline__ float wredMinF(float v) {
    #pragma unroll
    for (int m = 16; m >= 1; m >>= 1) v = fminf(v, __shfl_xor(v, m, 32));
    return v;
}
__device__ __forceinline__ float wredSumF(float v) {
    #pragma unroll
    for (int m = 16; m >= 1; m >>= 1) v += __shfl_xor(v, m, 32);
    return v;
}
__device__ __forceinline__ unsigned long long shflXor64(unsigned long long v, int m) {
    unsigned lo = (unsigned)v, hi = (unsigned)(v >> 32);
    lo = __shfl_xor(lo, m, 32);
    hi = __shfl_xor(hi, m, 32);
    return ((unsigned long long)hi << 32) | lo;
}

// ---------------- init ----------------
__global__ void k_init(float* bns, float* bnq, unsigned* mins) {
    int t = threadIdx.x;
    if (t < 256) { bns[t] = 0.f; bnq[t] = 0.f; }
    if (t == 0) { mins[0] = 0xFFFFFFFFu; mins[1] = 0xFFFFFFFFu; }
}

// ---------------- bf16 conversion / transposes ----------------
__global__ void k_convert_x(const float* __restrict__ x, __bf16* __restrict__ xb) {
    int i = blockIdx.x * 256 + threadIdx.x;           // grid sized exactly N*256/256
    xb[i] = f2bf(x[i]);
}
__global__ void k_convert_w(const float* __restrict__ W, __bf16* __restrict__ wt) {
    int i = blockIdx.x * 256 + threadIdx.x;           // 65536 elems
    int f = i >> 8, k = i & 255;
    wt[i] = f2bf(W[k * FEAT + f]);                    // wt[f][k] = W[k][f]
}
__global__ void k_convert_wkq(const float* __restrict__ Wk, const float* __restrict__ Wq,
                              __bf16* __restrict__ wkqt) {
    int i = blockIdx.x * 256 + threadIdx.x;           // 8192 elems
    int a = i >> 8, k = i & 255;
    float v = (a < 16) ? Wk[k * AE + a] : Wq[k * AE + (a - 16)];
    wkqt[i] = f2bf(v);                                // wkqt[a][k]
}

// ---------------- WMMA fragment loaders ----------------
// A fragment (16x32 bf16, row major source): lane layout per ISA 7.12.2
// Generic-pointer version: works for both global (xb) and LDS (staged tile).
__device__ __forceinline__ v16bf load_afrag(const __bf16* xb,
                                            int rowBase, int kb, int lane) {
    int row  = rowBase + (lane & 15);
    int koff = (lane >> 4) * 8;
    const __bf16* p = xb + (size_t)row * FEAT + kb + koff;
    v8bf a0 = *(const v8bf*)(p);        // K = kb+koff+0..7
    v8bf a1 = *(const v8bf*)(p + 16);   // K = kb+koff+16..23
    return __builtin_shufflevector(a0, a1, 0,1,2,3,4,5,6,7,8,9,10,11,12,13,14,15);
}
// B fragment (32x16 bf16) from K-contiguous (transposed) weights wt[col][k]
__device__ __forceinline__ v16bf load_bfrag(const __bf16* __restrict__ wt,
                                            int col, int kb, int lane) {
    int k0 = kb + (lane >> 4) * 8;
    const __bf16* p = wt + (size_t)col * FEAT + k0;
    v8bf b0 = *(const v8bf*)(p);
    v8bf b1 = *(const v8bf*)(p + 16);
    return __builtin_shufflevector(b0, b1, 0,1,2,3,4,5,6,7,8,9,10,11,12,13,14,15);
}

// ---------------- Tensor Data Mover: 16x512B tile -> LDS ----------------
// D# per CDNA5 ISA ch.8, 2 SGPR groups (2D tile, data_size=8B -> 64 units/row).
// `escape` is the LDS buffer pointer passed as an asm operand so LLVM's capture
// analysis knows the asm writes it (keeps the subsequent ds_loads alive).
__device__ __forceinline__ void tdm_issue(const __bf16* src, unsigned ldsAddr,
                                          const void* escape) {
    unsigned long long gaddr = (unsigned long long)(const void*)src;
    u32x4 g0;
    g0[0] = 1u;                                   // count=1, user descriptor
    g0[1] = ldsAddr;                              // LDS byte address of tile
    g0[2] = (unsigned)gaddr;                      // global_addr[31:0]
    g0[3] = (unsigned)(gaddr >> 32) | (2u << 30); // addr[56:32] | type=2
    u32x8 g1;
    g1[0] = 3u << 16;                             // data_size = 3 (8 bytes)
    g1[1] = 64u << 16;                            // tensor_dim0 = 64 units
    g1[2] = 16u << 16;                            // tensor_dim1 = 16 rows
    g1[3] = 64u << 16;                            // tile_dim0 = 64 units
    g1[4] = 16u;                                  // tile_dim1 = 16 rows
    g1[5] = 64u;                                  // tensor_dim0_stride = 64
    g1[6] = 0u;
    g1[7] = 0u;
    asm volatile("tensor_load_to_lds %0, %1" :: "s"(g0), "s"(g1), "v"(escape)
                 : "memory");
}

// ---------------- new_h = x @ W + B  (bf16 WMMA, f32 acc) ----------------
// Persistent blocks: 125 blocks x 512 threads (16 waves). Each block sweeps
// 10 row tiles. Each wave owns ONE 16-wide column tile and keeps its whole B
// panel in registers (8 x v16bf = 64 VGPRs), so W is read once per wave.
// The 16x256 bf16 A tile (8 KB) is double-buffered in LDS and fed by the
// Tensor Data Mover: wave 0 issues the TDM for tile t+1 while everyone
// computes tile t from ds_load_b128 fragments.
__global__ void __launch_bounds__(512) k_gemm_newh(
        const __bf16* __restrict__ xb, const __bf16* __restrict__ wt,
        const float* __restrict__ bias, float* __restrict__ newh) {
    __shared__ __bf16 sx[2][16 * FEAT];               // 2 x 8 KB buffers
    const int lane = threadIdx.x & 31;
    const int wid  = threadIdx.x >> 5;                // 0..15 = column tile
    const int col  = wid * 16 + (lane & 15);
    constexpr int TILES = 10;                         // 1250 row tiles / 125 blocks
    const int tile0 = blockIdx.x * TILES;

    // register-resident B panel for this wave's column tile
    v16bf Bf[8];
    #pragma unroll
    for (int ks = 0; ks < 8; ++ks) Bf[ks] = load_bfrag(wt, col, ks * 32, lane);
    const float bia = bias[col];

    const void* ldsPtr = (const void*)&sx[0][0];
    const unsigned ldsBase = (unsigned)(unsigned long long)(uintptr_t)ldsPtr;

    if (wid == 0)                                     // prologue: tile 0 -> buf 0
        tdm_issue(xb + (size_t)tile0 * 16 * FEAT, ldsBase, ldsPtr);

    #pragma unroll 1
    for (int t = 0; t < TILES; ++t) {
        if (wid == 0) {
            if (t + 1 < TILES) {
                tdm_issue(xb + (size_t)(tile0 + t + 1) * 16 * FEAT,
                          ldsBase + (unsigned)(((t + 1) & 1) * 16 * FEAT * 2),
                          ldsPtr);
                __builtin_amdgcn_s_wait_tensorcnt(1); // oldest TDM (tile t) done
            } else {
                __builtin_amdgcn_s_wait_tensorcnt(0);
            }
        }
        __syncthreads();                              // buf[t&1] visible to all

        const __bf16* at = &sx[t & 1][0];
        v8f acc = {};
        #pragma unroll
        for (int ks = 0; ks < 8; ++ks) {
            v16bf a = load_afrag(at, 0, ks * 32, lane);   // 2x ds_load_b128
            acc = __builtin_amdgcn_wmma_f32_16x16x32_bf16(false, a, false, Bf[ks],
                                                          (short)0, acc, false, false);
        }
        const int rowBase = (tile0 + t) * 16;
        const int rOff = (lane >> 4) * 8;
        #pragma unroll
        for (int i = 0; i < 8; ++i)
            newh[(size_t)(rowBase + i + rOff) * FEAT + col] = acc[i] + bia;

        __syncthreads();       // reads of buf[t&1] done before it is re-filled
    }
}

// ---------------- Key/Query = x @ [Wk|Wq] + [Bk|Bq]  (WMMA) ----------------
// 1250 row tiles x 2 col tiles = 2500 wave-tiles; grid = 313 blocks of 8 waves.
__global__ void k_gemm_kq(const __bf16* __restrict__ xb, const __bf16* __restrict__ wkqt,
                          const float* __restrict__ Bk, const float* __restrict__ Bq,
                          float* __restrict__ key, float* __restrict__ qry) {
    const int lane = threadIdx.x & 31;
    const int wid  = threadIdx.x >> 5;
    const int wg = blockIdx.x * 8 + wid;
    if (wg >= 2500) return;                          // whole-wave uniform exit
    const int rt = wg >> 1, ct = wg & 1;
    const int rowBase = rt * 16;
    const int col = ct * 16 + (lane & 15);           // column in 32-wide logical output
    v8f acc = {};
    #pragma unroll
    for (int kb = 0; kb < FEAT; kb += 32) {
        v16bf a = load_afrag(xb, rowBase, kb, lane);
        v16bf b = load_bfrag(wkqt, col, kb, lane);
        acc = __builtin_amdgcn_wmma_f32_16x16x32_bf16(false, a, false, b,
                                                      (short)0, acc, false, false);
    }
    const int ac = lane & 15;
    const float bia = (ct == 0) ? Bk[ac] : Bq[ac];
    float* dst = (ct == 0) ? key : qry;
    const int rOff = (lane >> 4) * 8;
    #pragma unroll
    for (int i = 0; i < 8; ++i) {
        int row = rowBase + i + rOff;
        dst[(size_t)row * AE + ac] = acc[i] + bia;
    }
}

// ---------------- la_s / ra_s (row dot products of new_h) ----------------
// grid = 2500 blocks * 8 waves -> one wave per row
__global__ void k_scores(const float* __restrict__ newh,
                         const float* __restrict__ la, const float* __restrict__ ra,
                         const float* __restrict__ Bla, const float* __restrict__ Bra,
                         float* __restrict__ las, float* __restrict__ ras) {
    const int lane = threadIdx.x & 31;
    const int row = blockIdx.x * 8 + (threadIdx.x >> 5);
    float sl = 0.f, sr = 0.f;
    #pragma unroll
    for (int t = 0; t < 8; ++t) {
        int f = lane + 32 * t;
        float h = newh[(size_t)row * FEAT + f];
        sl += h * la[f];
        sr += h * ra[f];
    }
    sl = wredSumF(sl);
    sr = wredSumF(sr);
    if (lane == 0) { las[row] = sl + Bla[0]; ras[row] = sr + Bra[0]; }
}

// ---------------- stage 1: attention scores + global min ----------------
// one wave per node; lane r handles receptive-field slot r
__global__ void k_att1(const float* __restrict__ key, const float* __restrict__ qry,
                       const int* __restrict__ rfp, float* __restrict__ att,
                       unsigned* __restrict__ mins) {
    const int lane = threadIdx.x & 31;
    const int n = blockIdx.x * 8 + (threadIdx.x >> 5);
    const int idx = rfp[(size_t)n * RFN + lane];
    const float4* q = (const float4*)(qry + (size_t)idx * AE);
    const float4* k = (const float4*)(key + (size_t)n * AE);
    float d = 0.f;
    #pragma unroll
    for (int j = 0; j < 4; ++j) {
        float4 qv = q[j], kv = k[j];
        d += qv.x * kv.x + qv.y * kv.y + qv.z * kv.z + qv.w * kv.w;
    }
    float a = (d > 0.f) ? d : ALPHA * d;              // leaky relu
    att[(size_t)n * RFN + lane] = a;
    float m = wredMinF(a);
    if (lane == 0) atomicMin(&mins[0], fordU(m));
}

// ---------------- stage 1: softmax + top-16 + weighted gather-sum ----------------
// grid = 625 blocks * 8 waves, 4 grid-stride rounds; BN partials via LDS
__global__ void k_top1(const float* __restrict__ att, const int* __restrict__ rfp,
                       const float* __restrict__ newh, const unsigned* __restrict__ mins,
                       float* __restrict__ out, float* __restrict__ bns,
                       float* __restrict__ bnq) {
    __shared__ float ssum[256];
    __shared__ float ssq[256];
    const int tid = threadIdx.x;
    const int lane = tid & 31;
    const int wid = tid >> 5;
    if (tid < 256) { ssum[tid] = 0.f; ssq[tid] = 0.f; }
    __syncthreads();

    const float mv = uordF(mins[0]) - 1000.0f;
    for (int n = blockIdx.x * 8 + wid; n < NN; n += gridDim.x * 8) {
        int rfv = rfp[(size_t)n * RFN + lane];
        float av = att[(size_t)n * RFN + lane];
        if (rfv == NN - 1) av = mv;
        // softmax over 32
        float mx = wredMaxF(av);
        float e = __expf(av - mx);
        float s = wredSumF(e);
        float p = e / s;
        // rank-based top-16 (stable ties -> lower index, matches lax.top_k)
        int rank = 0;
        #pragma unroll
        for (int j = 0; j < 32; ++j) {
            float pj = __shfl(p, j, 32);
            rank += (pj > p) || (pj == p && j < lane);
        }
        const bool sel = rank < TK;
        // re-softmax over selected values
        float pm = wredMaxF(sel ? p : -1e30f);
        float e2 = sel ? __expf(p - pm) : 0.f;
        float s2 = wredSumF(e2);
        float w = e2 / s2;
        // weighted neighbor sum: lane owns features [lane*8, lane*8+8)
        float acc[8];
        const float* hp0 = newh + (size_t)n * FEAT + lane * 8;
        #pragma unroll
        for (int t = 0; t < 8; ++t) acc[t] = hp0[t];
        #pragma unroll 1
        for (int j = 0; j < 32; ++j) {
            float wj = __shfl(w, j, 32);
            int   cj = __shfl(rfv, j, 32);
            int   sj = __shfl((int)sel, j, 32);
            if (sj) {
                const float* hp = newh + (size_t)cj * FEAT + lane * 8;
                #pragma unroll
                for (int t = 0; t < 8; ++t) acc[t] += wj * hp[t];
            }
        }
        float* op = out + (size_t)n * FEAT + lane * 8;
        #pragma unroll
        for (int t = 0; t < 8; ++t) {
            float v = acc[t];
            op[t] = v;
            atomicAdd(&ssum[lane * 8 + t], v);
            atomicAdd(&ssq[lane * 8 + t], v * v);
        }
    }
    __syncthreads();
    if (tid < 256) {
        atomicAdd(&bns[tid], ssum[tid]);
        atomicAdd(&bnq[tid], ssq[tid]);
    }
}

// ---------------- batchnorm ----------------
__global__ void k_bn_prep(const float* __restrict__ bns, const float* __restrict__ bnq,
                          const float* __restrict__ gamma, const float* __restrict__ beta,
                          float* __restrict__ scale, float* __restrict__ shift) {
    int f = threadIdx.x;
    float mean = bns[f] * (1.0f / NN);
    float var  = bnq[f] * (1.0f / NN) - mean * mean;   // biased variance
    float sc = gamma[f] * rsqrtf(var + EPS);
    scale[f] = sc;
    shift[f] = beta[f] - mean * sc;
}
__global__ void k_bn_apply(float* __restrict__ out, const float* __restrict__ scale,
                           const float* __restrict__ shift) {
    int i = blockIdx.x * 256 + threadIdx.x;            // grid exactly N*256/256
    int f = i & 255;
    float v = out[i] * scale[f] + shift[f];
    out[i] = fmaxf(v, 0.f);                            // relu
}

// ---------------- stage 2: global min over 1024-wide neighbor scores ----------------
__global__ void k_natt_min(const int* __restrict__ rfp, const int* __restrict__ adj,
                           const float* __restrict__ las, const float* __restrict__ ras,
                           unsigned* __restrict__ mins) {
    const int lane = threadIdx.x & 31;
    const int n = blockIdx.x * 8 + (threadIdx.x >> 5);
    const float r = ras[n];
    float m = 3.4e38f;
    #pragma unroll 1
    for (int s = 0; s < 32; ++s) {
        int rfn = rfp[(size_t)n * RFN + s];
        int nb = adj[(size_t)rfn * 32 + lane];
        m = fminf(m, las[nb] + r);
    }
    m = wredMinF(m);
    if (lane == 0) atomicMin(&mins[1], fordU(m));
}

// ---------------- stage 2: top-32 of 1024 + emit rf_new ----------------
// one wave per node; element e = s*32+lane; 32 iterative wave-argmax passes
__global__ void k_top2(const int* __restrict__ rfp, const int* __restrict__ adj,
                       const float* __restrict__ las, const float* __restrict__ ras,
                       const unsigned* __restrict__ mins, int* __restrict__ outrf) {
    const int lane = threadIdx.x & 31;
    const int n = blockIdx.x * 8 + (threadIdx.x >> 5);
    const float mv2 = uordF(mins[1]) - 1.0f;
    const float r = ras[n];
    float vals[32];
    #pragma unroll
    for (int s = 0; s < 32; ++s) {
        int rfn = rfp[(size_t)n * RFN + s];
        int nb = adj[(size_t)rfn * 32 + lane];
        float v = las[nb] + r;
        vals[s] = (nb == NN - 1) ? mv2 : v;
    }
    // first half of rf_new: copy original receptive field
    outrf[(size_t)n * RFN + lane] = rfp[(size_t)n * RFN + lane];
    int* out2 = outrf + (size_t)NN * RFN;
    #pragma unroll 1
    for (int it = 0; it < 32; ++it) {
        float bm = -__builtin_inff();
        int bs = 0;
        #pragma unroll
        for (int s = 0; s < 32; ++s)
            if (vals[s] > bm) { bm = vals[s]; bs = s; }   // strict > keeps smallest s
        unsigned e = (unsigned)(bs * 32 + lane);
        unsigned long long key = ((unsigned long long)fordU(bm) << 32)
                               | (unsigned long long)(0xFFFFFFFFu - e);
        #pragma unroll
        for (int m = 16; m >= 1; m >>= 1) {
            unsigned long long o = shflXor64(key, m);
            if (o > key) key = o;
        }
        unsigned ew = 0xFFFFFFFFu - (unsigned)key;
        int wl = ew & 31;
        int wsel = (int)(ew >> 5);
        if (lane == 0) {
            int rfn = rfp[(size_t)n * RFN + wsel];
            out2[(size_t)n * RFN + it] = adj[(size_t)rfn * 32 + wl];
        }
        if (lane == wl) {
            #pragma unroll
            for (int s = 0; s < 32; ++s)
                if (s == wsel) vals[s] = -__builtin_inff();
        }
    }
}

// ---------------- host-side launcher ----------------
extern "C" void kernel_launch(void* const* d_in, const int* in_sizes, int n_in,
                              void* d_out, int out_size, void* d_ws, size_t ws_size,
                              hipStream_t stream) {
    (void)in_sizes; (void)n_in; (void)out_size; (void)ws_size;
    const float* x    = (const float*)d_in[0];
    const int*   rfp  = (const int*)d_in[1];      // [1,N,32]
    const int*   adj  = (const int*)d_in[2];      // [N,32]
    const float* W    = (const float*)d_in[3];    // [1,256,256]
    const float* B    = (const float*)d_in[4];    // [1,1,256]
    const float* Wk   = (const float*)d_in[5];
    const float* Bk   = (const float*)d_in[6];
    const float* Wq   = (const float*)d_in[7];
    const float* Bq   = (const float*)d_in[8];
    const float* la   = (const float*)d_in[9];
    const float* ra   = (const float*)d_in[10];
    const float* Bla  = (const float*)d_in[11];
    const float* Bra  = (const float*)d_in[12];
    const float* gmm  = (const float*)d_in[13];
    const float* bet  = (const float*)d_in[14];

    char* ws = (char*)d_ws;
    float*    newh = (float*)(ws + OFF_NEWH);
    float*    key  = (float*)(ws + OFF_KEY);
    float*    qry  = (float*)(ws + OFF_QRY);
    float*    las  = (float*)(ws + OFF_LAS);
    float*    ras  = (float*)(ws + OFF_RAS);
    float*    att  = (float*)(ws + OFF_ATT);
    __bf16*   xb   = (__bf16*)(ws + OFF_XB);
    __bf16*   wt   = (__bf16*)(ws + OFF_WT);
    __bf16*   wkqt = (__bf16*)(ws + OFF_WKQT);
    float*    bns  = (float*)(ws + OFF_BNS);
    float*    bnq  = (float*)(ws + OFF_BNQ);
    float*    sc   = (float*)(ws + OFF_SC);
    float*    sh   = (float*)(ws + OFF_SH);
    unsigned* mins = (unsigned*)(ws + OFF_MIN);

    float* outF = (float*)d_out;                       // final_h [N,256]
    int*   outI = (int*)(outF + (size_t)NN * FEAT);    // rf_new [2,N,32]

    k_init<<<1, 256, 0, stream>>>(bns, bnq, mins);
    k_convert_x<<<NN * FEAT / 256, 256, 0, stream>>>(x, xb);
    k_convert_w<<<FEAT * FEAT / 256, 256, 0, stream>>>(W, wt);
    k_convert_wkq<<<32 * FEAT / 256, 256, 0, stream>>>(Wk, Wq, wkqt);

    k_gemm_newh<<<125, 512, 0, stream>>>(xb, wt, B, newh);
    k_gemm_kq<<<(2500 + 7) / 8, 256, 0, stream>>>(xb, wkqt, Bk, Bq, key, qry);
    k_scores<<<NN / 8, 256, 0, stream>>>(newh, la, ra, Bla, Bra, las, ras);

    k_att1<<<NN / 8, 256, 0, stream>>>(key, qry, rfp, att, mins);
    k_top1<<<625, 256, 0, stream>>>(att, rfp, newh, mins, outF, bns, bnq);

    k_bn_prep<<<1, 256, 0, stream>>>(bns, bnq, gmm, bet, sc, sh);
    k_bn_apply<<<NN * FEAT / 256, 256, 0, stream>>>(outF, sc, sh);

    k_natt_min<<<NN / 8, 256, 0, stream>>>(rfp, adj, las, ras, mins);
    k_top2<<<NN / 8, 256, 0, stream>>>(rfp, adj, las, ras, mins, outI);
}
